// DecoderRNN_42021960024657
// MI455X (gfx1250) — compile-verified, matrix operand order checked by reading, not run on hardware
//
#include <hip/hip_runtime.h>

// ---------------------------------------------------------------------------
// Types for gfx1250 WMMA (wave32): D(16x16 f32) = A(16x32 bf16) x B(32x16 bf16) + C
// ---------------------------------------------------------------------------
typedef __attribute__((ext_vector_type(16))) __bf16 v16bf;
typedef __attribute__((ext_vector_type(8)))  float  v8f;

#define BATCH 64
#define HID   1024
#define EMB   512
#define VOCAB 10000
#define TSTEP 256
#define SOS_ID 1

__device__ inline v8f wmma_bf16(v16bf a, v16bf b, v8f c) {
  return __builtin_amdgcn_wmma_f32_16x16x32_bf16(
      /*neg_a=*/false, a, /*neg_b=*/false, b,
      /*c_mod=*/(short)0, c, /*reuse_a=*/false, /*reuse_b=*/false);
}

// Load a 16x32 bf16 fragment (A-layout; B is the mirrored layout with the same
// per-lane K ordering: column n of B == row `col` of row-major W).
// Lane L: row = row0 + (L&15); khalf = L>>4.
// Fragment = dwords [4*kh .. 4*kh+3] and [8+4*kh .. 8+4*kh+3] of the row slice
// -> exactly two 16B-aligned global_load_b128.
__device__ inline v16bf load_frag(const __bf16* __restrict__ base, int row0,
                                  int ld, int k0, int lane) {
  union { v16bf v; uint4 q[2]; } f;
  const int r  = row0 + (lane & 15);
  const int kh = lane >> 4;
  const uint4* p =
      reinterpret_cast<const uint4*>(base + (size_t)r * ld + k0);
  f.q[0] = p[kh];
  f.q[1] = p[2 + kh];
  return f.v;
}

__device__ inline float sigmoidf_(float x) { return 1.0f / (1.0f + expf(-x)); }

// ---------------------------------------------------------------------------
// One-time setup kernels
// ---------------------------------------------------------------------------
__global__ void cvt_bf16_kernel(const float* __restrict__ s,
                                __bf16* __restrict__ d, int n) {
  int i = blockIdx.x * blockDim.x + threadIdx.x;
  if (i < n) d[i] = (__bf16)s[i];
}

__global__ void init_state_kernel(const float* __restrict__ enc,
                                  float* __restrict__ h0f, __bf16* __restrict__ h0b,
                                  float* __restrict__ h1f, __bf16* __restrict__ h1b) {
  int i = blockIdx.x * blockDim.x + threadIdx.x;  // 2*64*1024
  if (i >= 2 * BATCH * HID) return;
  const int l = i >> 16;          // 65536 per layer
  const int j = i & 0xFFFF;
  const float v = enc[i];
  if (l == 0) { h0f[j] = v; h0b[j] = (__bf16)v; }
  else        { h1f[j] = v; h1b[j] = (__bf16)v; }
}

__global__ void init_tok_kernel(int* __restrict__ tok) {
  int i = threadIdx.x;
  if (i < BATCH) tok[i] = SOS_ID;
}

// ---------------------------------------------------------------------------
// Per-step kernels
// ---------------------------------------------------------------------------
__global__ void embed_relu_kernel(const float* __restrict__ embW,
                                  const int* __restrict__ tok,
                                  __bf16* __restrict__ x) {
  int i = blockIdx.x * blockDim.x + threadIdx.x;  // 64*512
  if (i >= BATCH * EMB) return;
  const int b = i >> 9;
  const int e = i & (EMB - 1);
  float v = embW[(size_t)tok[b] * EMB + e];
  x[i] = (__bf16)(v > 0.0f ? v : 0.0f);
}

// Fused GRU cell. Block = 8 waves, computes a 64x64 tile of h_new.
// Wave w: rows [16*(w&3), +16), cols cA = bid*64 + (w>>2)*32 and cB = cA+16.
// r/z accumulators take both x@Wih^T and h@Whh^T contributions.
// All 7 fragment loads are issued before the 6 WMMAs so the scheduler can
// overlap L2 latency with matrix issue (partial loadcnt waits).
__global__ __launch_bounds__(256) void gru_step_kernel(
    const __bf16* __restrict__ x, int Kx,
    const float* __restrict__ h, const __bf16* __restrict__ hb,
    const __bf16* __restrict__ Wih, const __bf16* __restrict__ Whh,
    const float* __restrict__ bih, const float* __restrict__ bhh,
    float* __restrict__ h_out, __bf16* __restrict__ hb_out) {
  const int H = HID;
  const int wave = threadIdx.x >> 5;
  const int lane = threadIdx.x & 31;
  const int m0 = (wave & 3) * 16;
  const int cA = blockIdx.x * 64 + (wave >> 2) * 32;
  const int cB = cA + 16;

  v8f ar0 = {}, ar1 = {}, az0 = {}, az1 = {};
  v8f ain0 = {}, ain1 = {}, ahn0 = {}, ahn1 = {};

  // gi = x @ Wih^T   (gates r,z,n at row offsets 0,H,2H of Wih)
#pragma unroll 1
  for (int k = 0; k < Kx; k += 32) {
    v16bf a  = load_frag(x,   m0,        Kx, k, lane);
    v16bf b0 = load_frag(Wih, cA,        Kx, k, lane);
    v16bf b1 = load_frag(Wih, cB,        Kx, k, lane);
    v16bf b2 = load_frag(Wih, H + cA,    Kx, k, lane);
    v16bf b3 = load_frag(Wih, H + cB,    Kx, k, lane);
    v16bf b4 = load_frag(Wih, 2 * H + cA, Kx, k, lane);
    v16bf b5 = load_frag(Wih, 2 * H + cB, Kx, k, lane);
    ar0  = wmma_bf16(a, b0, ar0);
    ar1  = wmma_bf16(a, b1, ar1);
    az0  = wmma_bf16(a, b2, az0);
    az1  = wmma_bf16(a, b3, az1);
    ain0 = wmma_bf16(a, b4, ain0);
    ain1 = wmma_bf16(a, b5, ain1);
  }
  // gh = h @ Whh^T  (r,z share accumulators with gi; n kept separate)
#pragma unroll 1
  for (int k = 0; k < H; k += 32) {
    v16bf a  = load_frag(hb,  m0,        H, k, lane);
    v16bf b0 = load_frag(Whh, cA,        H, k, lane);
    v16bf b1 = load_frag(Whh, cB,        H, k, lane);
    v16bf b2 = load_frag(Whh, H + cA,    H, k, lane);
    v16bf b3 = load_frag(Whh, H + cB,    H, k, lane);
    v16bf b4 = load_frag(Whh, 2 * H + cA, H, k, lane);
    v16bf b5 = load_frag(Whh, 2 * H + cB, H, k, lane);
    ar0  = wmma_bf16(a, b0, ar0);
    ar1  = wmma_bf16(a, b1, ar1);
    az0  = wmma_bf16(a, b2, az0);
    az1  = wmma_bf16(a, b3, az1);
    ahn0 = wmma_bf16(a, b4, ahn0);
    ahn1 = wmma_bf16(a, b5, ahn1);
  }

  // Epilogue: gates + state update, all in registers.
  const int rbase = m0 + ((lane >> 4) ? 8 : 0);
  const int nlo = lane & 15;
#pragma unroll
  for (int p = 0; p < 8; ++p) {
    const int row = rbase + p;
    {
      const int col = cA + nlo;
      float rr = sigmoidf_(ar0[p] + bih[col] + bhh[col]);
      float zz = sigmoidf_(az0[p] + bih[H + col] + bhh[H + col]);
      float nn = tanhf(ain0[p] + bih[2 * H + col] + rr * (ahn0[p] + bhh[2 * H + col]));
      float o = (1.0f - zz) * nn + zz * h[row * H + col];
      h_out[row * H + col] = o;
      hb_out[row * H + col] = (__bf16)o;
    }
    {
      const int col = cB + nlo;
      float rr = sigmoidf_(ar1[p] + bih[col] + bhh[col]);
      float zz = sigmoidf_(az1[p] + bih[H + col] + bhh[H + col]);
      float nn = tanhf(ain1[p] + bih[2 * H + col] + rr * (ahn1[p] + bhh[2 * H + col]));
      float o = (1.0f - zz) * nn + zz * h[row * H + col];
      h_out[row * H + col] = o;
      hb_out[row * H + col] = (__bf16)o;
    }
  }
}

// hid = relu(h1n @ pW1^T + pb1), stored as bf16 (GEMM input only).
__global__ __launch_bounds__(256) void proj_relu_kernel(
    const __bf16* __restrict__ x, const __bf16* __restrict__ W,
    const float* __restrict__ bias, __bf16* __restrict__ out) {
  const int K = HID;
  const int wave = threadIdx.x >> 5;
  const int lane = threadIdx.x & 31;
  const int m0 = (wave & 3) * 16;
  const int cA = blockIdx.x * 64 + (wave >> 2) * 32;
  const int cB = cA + 16;
  v8f a0 = {}, a1 = {};
#pragma unroll 1
  for (int k = 0; k < K; k += 32) {
    v16bf a  = load_frag(x, m0, K, k, lane);
    v16bf b0 = load_frag(W, cA, K, k, lane);
    v16bf b1 = load_frag(W, cB, K, k, lane);
    a0 = wmma_bf16(a, b0, a0);
    a1 = wmma_bf16(a, b1, a1);
  }
  const int rbase = m0 + ((lane >> 4) ? 8 : 0);
  const int nlo = lane & 15;
#pragma unroll
  for (int p = 0; p < 8; ++p) {
    const int row = rbase + p;
    float v0 = a0[p] + bias[cA + nlo];
    float v1 = a1[p] + bias[cB + nlo];
    out[row * HID + cA + nlo] = (__bf16)(v0 > 0.0f ? v0 : 0.0f);
    out[row * HID + cB + nlo] = (__bf16)(v1 > 0.0f ? v1 : 0.0f);
  }
}

// logits[t] = hid @ pW2^T + pb2, written into d_out[b][t][:]
__global__ __launch_bounds__(256) void logits_kernel(
    const __bf16* __restrict__ x, const __bf16* __restrict__ W,
    const float* __restrict__ bias, float* __restrict__ out_t,
    long long row_stride) {
  const int K = HID;
  const int wave = threadIdx.x >> 5;
  const int lane = threadIdx.x & 31;
  const int m0 = (wave & 3) * 16;
  const int cA = blockIdx.x * 64 + (wave >> 2) * 32;
  const int cB = cA + 16;
  const bool vA = (cA < VOCAB);  // VOCAB % 16 == 0 -> tiles fully valid/invalid
  const bool vB = (cB < VOCAB);
  v8f a0 = {}, a1 = {};
#pragma unroll 1
  for (int k = 0; k < K; k += 32) {
    v16bf a = load_frag(x, m0, K, k, lane);
    if (vA) {
      v16bf b0 = load_frag(W, cA, K, k, lane);
      if (vB) {
        v16bf b1 = load_frag(W, cB, K, k, lane);
        a0 = wmma_bf16(a, b0, a0);
        a1 = wmma_bf16(a, b1, a1);
      } else {
        a0 = wmma_bf16(a, b0, a0);
      }
    }
  }
  const int rbase = m0 + ((lane >> 4) ? 8 : 0);
  const int nlo = lane & 15;
#pragma unroll
  for (int p = 0; p < 8; ++p) {
    const int row = rbase + p;
    if (vA) out_t[(size_t)row * row_stride + cA + nlo] = a0[p] + bias[cA + nlo];
    if (vB) out_t[(size_t)row * row_stride + cB + nlo] = a1[p] + bias[cB + nlo];
  }
}

// First-max argmax over VOCAB per batch row (matches jnp.argmax tie-breaking).
__global__ __launch_bounds__(256) void argmax_kernel(
    const float* __restrict__ logits_t, long long row_stride,
    int* __restrict__ tok) {
  __shared__ float sv[256];
  __shared__ int   si[256];
  const int b = blockIdx.x;
  const float* p = logits_t + (size_t)b * row_stride;
  float bv = -__builtin_inff();
  int bi = 0x7fffffff;
  for (int v = threadIdx.x; v < VOCAB; v += 256) {
    float f = p[v];
    if (f > bv || (f == bv && v < bi)) { bv = f; bi = v; }
  }
  sv[threadIdx.x] = bv; si[threadIdx.x] = bi;
  __syncthreads();
  for (int s = 128; s > 0; s >>= 1) {
    if (threadIdx.x < s) {
      float ov = sv[threadIdx.x + s]; int oi = si[threadIdx.x + s];
      if (ov > sv[threadIdx.x] || (ov == sv[threadIdx.x] && oi < si[threadIdx.x])) {
        sv[threadIdx.x] = ov; si[threadIdx.x] = oi;
      }
    }
    __syncthreads();
  }
  if (threadIdx.x == 0) tok[b] = si[0];
}

// ---------------------------------------------------------------------------
// Host launcher
// ---------------------------------------------------------------------------
extern "C" void kernel_launch(void* const* d_in, const int* in_sizes, int n_in,
                              void* d_out, int out_size, void* d_ws, size_t ws_size,
                              hipStream_t stream) {
  (void)in_sizes; (void)n_in; (void)out_size; (void)ws_size;
  // Inputs (setup_inputs order)
  const float* enc_final = (const float*)d_in[1];   // (2,64,1024)
  const float* emb_W     = (const float*)d_in[2];   // (10000,512)
  const float* Wih0      = (const float*)d_in[3];   // (3072,512)
  const float* Whh0      = (const float*)d_in[4];   // (3072,1024)
  const float* bih0      = (const float*)d_in[5];
  const float* bhh0      = (const float*)d_in[6];
  const float* Wih1      = (const float*)d_in[7];   // (3072,1024)
  const float* Whh1      = (const float*)d_in[8];
  const float* bih1      = (const float*)d_in[9];
  const float* bhh1      = (const float*)d_in[10];
  const float* pW1       = (const float*)d_in[11];  // (1024,1024)
  const float* pb1       = (const float*)d_in[12];
  const float* pW2       = (const float*)d_in[13];  // (10000,1024)
  const float* pb2       = (const float*)d_in[14];
  float* out = (float*)d_out;                       // (64,256,10000)

  // Workspace carve-up (256B aligned)
  char* ws = (char*)d_ws;
  size_t off = 0;
  auto alloc = [&](size_t bytes) -> char* {
    char* p = ws + off;
    off = (off + bytes + 255) & ~(size_t)255;
    return p;
  };
  __bf16* wih0_b = (__bf16*)alloc((size_t)3 * HID * EMB * 2);
  __bf16* whh0_b = (__bf16*)alloc((size_t)3 * HID * HID * 2);
  __bf16* wih1_b = (__bf16*)alloc((size_t)3 * HID * HID * 2);
  __bf16* whh1_b = (__bf16*)alloc((size_t)3 * HID * HID * 2);
  __bf16* pw1_b  = (__bf16*)alloc((size_t)HID * HID * 2);
  __bf16* pw2_b  = (__bf16*)alloc((size_t)VOCAB * HID * 2);
  __bf16* x_b    = (__bf16*)alloc((size_t)BATCH * EMB * 2);
  float*  h0f[2]; __bf16* h0b[2]; float* h1f[2]; __bf16* h1b[2];
  for (int i = 0; i < 2; ++i) {
    h0f[i] = (float*)alloc((size_t)BATCH * HID * 4);
    h0b[i] = (__bf16*)alloc((size_t)BATCH * HID * 2);
    h1f[i] = (float*)alloc((size_t)BATCH * HID * 4);
    h1b[i] = (__bf16*)alloc((size_t)BATCH * HID * 2);
  }
  __bf16* hid_b = (__bf16*)alloc((size_t)BATCH * HID * 2);
  int* tok = (int*)alloc((size_t)BATCH * 4);

  // --- One-time weight conversion (fp32 -> bf16, resident in L2 thereafter)
  auto cvt = [&](const float* s, __bf16* d, int n) {
    cvt_bf16_kernel<<<(n + 255) / 256, 256, 0, stream>>>(s, d, n);
  };
  cvt(Wih0, wih0_b, 3 * HID * EMB);
  cvt(Whh0, whh0_b, 3 * HID * HID);
  cvt(Wih1, wih1_b, 3 * HID * HID);
  cvt(Whh1, whh1_b, 3 * HID * HID);
  cvt(pW1,  pw1_b,  HID * HID);
  cvt(pW2,  pw2_b,  VOCAB * HID);

  init_state_kernel<<<(2 * BATCH * HID + 255) / 256, 256, 0, stream>>>(
      enc_final, h0f[0], h0b[0], h1f[0], h1b[0]);
  init_tok_kernel<<<1, 64, 0, stream>>>(tok);

  const long long row_stride = (long long)TSTEP * VOCAB;
  const int n_blk_h = HID / 64;                    // 16
  const int n_blk_v = (VOCAB + 63) / 64;           // 157

  for (int t = 0; t < TSTEP; ++t) {
    const int r = t & 1, w = r ^ 1;
    embed_relu_kernel<<<(BATCH * EMB + 255) / 256, 256, 0, stream>>>(emb_W, tok, x_b);
    gru_step_kernel<<<n_blk_h, 256, 0, stream>>>(
        x_b, EMB, h0f[r], h0b[r], wih0_b, whh0_b, bih0, bhh0, h0f[w], h0b[w]);
    gru_step_kernel<<<n_blk_h, 256, 0, stream>>>(
        h0b[w], HID, h1f[r], h1b[r], wih1_b, whh1_b, bih1, bhh1, h1f[w], h1b[w]);
    proj_relu_kernel<<<n_blk_h, 256, 0, stream>>>(h1b[w], pw1_b, pb1, hid_b);
    float* out_t = out + (size_t)t * VOCAB;
    logits_kernel<<<n_blk_v, 256, 0, stream>>>(hid_b, pw2_b, pb2, out_t, row_stride);
    argmax_kernel<<<BATCH, 256, 0, stream>>>(out_t, row_stride, tok);
  }
}
